// KGramEmbeddingSeqModel_17274358464626
// MI455X (gfx1250) — matrix operand order repaired and checked by reading, not compiled, not run on hardware
//
#include <hip/hip_runtime.h>
#include <cstdint>
#include <cstddef>

// ---------------------------------------------------------------------------
// KGram embedding + MLP (silu) + vocab projection for MI455X (gfx1250).
//   - GEMMs on v_wmma_f32_16x16x32_bf16 (bf16 in, f32 accumulate)
//   - GEMM2 B-stream staged through LDS by the Tensor Data Mover
//     (tensor_load_to_lds + s_wait_tensorcnt), double buffered.
// ---------------------------------------------------------------------------

typedef __attribute__((ext_vector_type(16))) __bf16 v16bf;
typedef __attribute__((ext_vector_type(8)))  float  v8f;
typedef unsigned int u32x4 __attribute__((ext_vector_type(4)));
typedef int          i32x4 __attribute__((ext_vector_type(4)));
typedef int          i32x8 __attribute__((ext_vector_type(8)));

#define S_LEN   1024
#define BATCH   2
#define M_ROWS  2048          // S*B
#define D_EMB   341
#define KCTX    3
#define HID     1023
#define KP      1024          // K padded to multiple of 32
#define VOCAB   50257
#define VPP     50432         // vocab padded to 256*197 (full block tiles)
// GEMM2 block tiling: 64(M) x 256(N) per block, 8 waves in 2x4 grid,
// each wave: 2(M) x 4(N) 16x16 tiles.
#define BM      64
#define BN      256
#define G2_MWT  2
#define G2_NWT  4
// GEMM1 per-wave tiling (direct-load kernel)
#define MWT     2
#define NWT     4

// round-to-nearest-even fp32 -> bf16 (raw u16)
__device__ __forceinline__ unsigned short f2bf(float f) {
  unsigned int u = __float_as_uint(f);
  unsigned int r = (u + 0x7FFFu + ((u >> 16) & 1u)) >> 16;
  return (unsigned short)r;
}

__device__ __forceinline__ float silu(float x) {
  return x / (1.0f + __expf(-x));
}

union FragU { uint4 q[2]; v16bf v; };

// A fragment for v_wmma_f32_16x16x32_bf16 (16-bit A 16x32):
// lane half 0 holds K 0..15, half 1 holds K 16..31 (matches B fragment).
__device__ __forceinline__ v16bf load_a_frag(const unsigned short* row_k0, int half) {
  FragU u;
  u.q[0] = *reinterpret_cast<const uint4*>(row_k0 + half * 16);
  u.q[1] = *reinterpret_cast<const uint4*>(row_k0 + half * 16 + 8);
  return u.v;
}

// B fragment: lane n = lane%16 is output column, halves take K 0..15 / 16..31
// sequentially from K-contiguous storage.
__device__ __forceinline__ v16bf load_b_frag(const unsigned short* col_k0, int half) {
  FragU u;
  const uint4* q = reinterpret_cast<const uint4*>(col_k0 + half * 16);
  u.q[0] = q[0];
  u.q[1] = q[1];
  return u.v;
}

// ---------------------------------------------------------------------------
// TDM: issue a 2D tensor_load_to_lds (bf16 elements).
// Tile = tile_d0 (contiguous K elems) x tile_d1 (rows); row stride = stride_d0
// elements. D# built per CDNA5 ISA ch.8 (group0/group1; groups 2,3 unused).
// ---------------------------------------------------------------------------
#if __has_builtin(__builtin_amdgcn_tensor_load_to_lds)
#define HAVE_TDM 1
__device__ __forceinline__ void tdm_load_2d(unsigned lds_off, const void* gaddr,
                                            unsigned tensor_d0, unsigned tensor_d1,
                                            unsigned long long stride_d0,
                                            unsigned tile_d0, unsigned tile_d1) {
  unsigned long long ga = (unsigned long long)gaddr;
  u32x4 g0;
  g0[0] = 1u;                                               // count=1, user D#
  g0[1] = lds_off;                                          // lds_addr (bytes)
  g0[2] = (unsigned)ga;                                     // global_addr lo
  g0[3] = (unsigned)((ga >> 32) & 0x1FFFFFFu) | (2u << 30); // addr hi | type=2
  i32x8 g1;
  g1[0] = (int)(1u << 16);                                  // data_size=1 (2B)
  g1[1] = (int)((tensor_d0 & 0xFFFFu) << 16);               // tensor_dim0 lo16
  g1[2] = (int)((tensor_d0 >> 16) | ((tensor_d1 & 0xFFFFu) << 16));
  g1[3] = (int)((tensor_d1 >> 16) | (tile_d0 << 16));       // | tile_dim0
  g1[4] = (int)(tile_d1 & 0xFFFFu);                         // tile_dim1, dim2=0
  g1[5] = (int)(unsigned)(stride_d0 & 0xFFFFFFFFull);       // dim0 stride lo
  g1[6] = (int)(unsigned)((stride_d0 >> 32) & 0xFFFFull);   // stride hi, d1str=0
  g1[7] = 0;
  i32x4 z4 = (i32x4)0;
#if __clang_major__ >= 23
  i32x8 z8 = (i32x8)0;
  __builtin_amdgcn_tensor_load_to_lds(g0, g1, z4, z4, z8, 0);
#else
  __builtin_amdgcn_tensor_load_to_lds(g0, g1, z4, z4, 0);
#endif
}
#else
#define HAVE_TDM 0
#endif

// ---------------------------------------------------------------------------
// Kernel 1: embedding gather with the reference's raw (K,B,D)->(B,K*D)
// reshape semantics, emitted as bf16 into Xb[2048][1024] (col 1023 zero pad).
// ---------------------------------------------------------------------------
__global__ void pack_x_kernel(const int* __restrict__ tokens,     // (S,B)
                              const float* __restrict__ embed,    // (VOCAB,D)
                              unsigned short* __restrict__ Xb) {  // (2048,1024)
  int idx = blockIdx.x * blockDim.x + threadIdx.x;
  int row = idx >> 10;
  int j   = idx & 1023;
  if (row >= M_ROWS) return;
  float val = 0.0f;
  if (j < HID) {
    int s  = row >> 1;
    int bp = row & 1;
    int flat = bp * HID + j;            // raw flatten of (K,B,D)
    int k   = flat / (BATCH * D_EMB);
    int rem = flat % (BATCH * D_EMB);
    int b   = rem / D_EMB;
    int d   = rem % D_EMB;
    int t   = s + k;
    int tok = (t < KCTX) ? 0 : tokens[(t - KCTX) * BATCH + b];
    val = embed[(size_t)tok * D_EMB + d];
  }
  Xb[(size_t)row * KP + j] = f2bf(val);
}

// ---------------------------------------------------------------------------
// Kernel 2: tiled transpose + fp32->bf16: W (Krows x Ncols) -> Wt (NP x KP),
// Wt[n][k] = W[k][n], zero padded; coalesced both sides via 32x33 LDS tile.
// ---------------------------------------------------------------------------
__global__ void transpose_bf16_kernel(const float* __restrict__ W,
                                      unsigned short* __restrict__ Wt,
                                      int Krows, int Ncols, int NProws) {
  __shared__ float tile[32][33];
  int bk = blockIdx.x * 32;
  int bn = blockIdx.y * 32;
  int tx = threadIdx.x;
  int ty = threadIdx.y;
#pragma unroll
  for (int i = ty; i < 32; i += 8) {
    int k = bk + i, n = bn + tx;
    tile[i][tx] = (k < Krows && n < Ncols) ? W[(size_t)k * Ncols + n] : 0.0f;
  }
  __syncthreads();
#pragma unroll
  for (int i = ty; i < 32; i += 8) {
    int n = bn + i, k = bk + tx;
    if (n < NProws && k < KP)
      Wt[(size_t)n * KP + k] = f2bf(tile[tx][i]);
  }
}

// ---------------------------------------------------------------------------
// Kernel 3: GEMM1  H = silu(X @ W1 + b1) -> bf16 (col 1023 zeroed).
// Direct-load WMMA kernel (A and B both hot in L2; only 4.3 GFLOP).
// ---------------------------------------------------------------------------
__global__ void gemm1_silu_kernel(const unsigned short* __restrict__ Xb,
                                  const unsigned short* __restrict__ W1t,
                                  const float* __restrict__ b1,
                                  unsigned short* __restrict__ Hb,
                                  int ngroups_n) {
  const int lane = threadIdx.x & 31;
  const int wave = threadIdx.x >> 5;
  const int n16  = lane & 15;
  const int half = lane >> 4;
  const int wid  = blockIdx.x * 8 + wave;
  const int mg   = wid / ngroups_n;
  const int ng   = wid % ngroups_n;
  const int m0   = mg * (16 * MWT);
  const int n0   = ng * (16 * NWT);

  const unsigned short* arow[MWT];
#pragma unroll
  for (int mi = 0; mi < MWT; ++mi)
    arow[mi] = Xb + (size_t)(m0 + mi * 16 + n16) * KP;
  const unsigned short* bcol[NWT];
#pragma unroll
  for (int ni = 0; ni < NWT; ++ni)
    bcol[ni] = W1t + (size_t)(n0 + ni * 16 + n16) * KP;

  v8f acc[MWT][NWT];
#pragma unroll
  for (int mi = 0; mi < MWT; ++mi)
#pragma unroll
    for (int ni = 0; ni < NWT; ++ni)
      acc[mi][ni] = (v8f)(0.0f);

  for (int k0 = 0; k0 < KP; k0 += 32) {
    v16bf a[MWT];
#pragma unroll
    for (int mi = 0; mi < MWT; ++mi)
      a[mi] = load_a_frag(arow[mi] + k0, half);
#pragma unroll
    for (int ni = 0; ni < NWT; ++ni) {
      v16bf b = load_b_frag(bcol[ni] + k0, half);
#pragma unroll
      for (int mi = 0; mi < MWT; ++mi)
        acc[mi][ni] = __builtin_amdgcn_wmma_f32_16x16x32_bf16(
            false, a[mi], false, b, (short)0, acc[mi][ni], false, false);
    }
  }

#pragma unroll
  for (int ni = 0; ni < NWT; ++ni) {
    int col = n0 + ni * 16 + n16;
    float bias = (col < HID) ? b1[col] : 0.0f;
#pragma unroll
    for (int mi = 0; mi < MWT; ++mi) {
#pragma unroll
      for (int r = 0; r < 8; ++r) {
        int row = m0 + mi * 16 + 8 * half + r;
        float v = (col < HID) ? silu(acc[mi][ni][r] + bias) : 0.0f;
        Hb[(size_t)row * KP + col] = f2bf(v);
      }
    }
  }
}

// ---------------------------------------------------------------------------
// Kernel 4: GEMM2  out = H @ W2 + b2 (fp32 out).
// Block: 64(M) x 256(N). Wave w (0..7): wm = w>>2 (M pair), wn = w&3 (N quad).
// B (W2t) tiles staged in LDS by the Tensor Data Mover, double buffered:
// wave 0 issues tensor_load_to_lds for the next 256x32 bf16 chunk while all
// 8 waves run WMMA out of the current one; sync = s_wait_tensorcnt + barrier.
// ---------------------------------------------------------------------------
__global__ void gemm2_kernel(const unsigned short* __restrict__ Hb,   // (2048,1024)
                             const unsigned short* __restrict__ W2t,  // (VPP,1024)
                             const float* __restrict__ b2,            // (VOCAB)
                             float* __restrict__ out) {               // (2048,VOCAB)
  __shared__ unsigned short sB[2][BN * 32];   // 2 x 16 KB

  const int lane = threadIdx.x & 31;
  const int wave = threadIdx.x >> 5;
  const int n16  = lane & 15;
  const int half = lane >> 4;
  const int wm   = wave >> 2;      // 0..1
  const int wn   = wave & 3;       // 0..3
  const int m0   = blockIdx.x * BM;
  const int n0   = blockIdx.y * BN;

  const unsigned short* arow[G2_MWT];
#pragma unroll
  for (int mi = 0; mi < G2_MWT; ++mi)
    arow[mi] = Hb + (size_t)(m0 + wm * 32 + mi * 16 + n16) * KP;
  const unsigned short* bsrc = W2t + (size_t)n0 * KP;   // chunk k0 added later

  v8f acc[G2_MWT][G2_NWT];
#pragma unroll
  for (int mi = 0; mi < G2_MWT; ++mi)
#pragma unroll
    for (int ni = 0; ni < G2_NWT; ++ni)
      acc[mi][ni] = (v8f)(0.0f);

  const int NCHUNK = KP / 32;   // 32

#if HAVE_TDM
  if (threadIdx.x < 32)
    tdm_load_2d((unsigned)(size_t)&sB[0][0], bsrc,
                /*tensor_d0=*/KP, /*tensor_d1=*/VPP,
                /*stride_d0=*/(unsigned long long)KP,
                /*tile_d0=*/32, /*tile_d1=*/BN);

  for (int kc = 0; kc < NCHUNK; ++kc) {
    const int buf = kc & 1;
    if (threadIdx.x < 32)
      __builtin_amdgcn_s_wait_tensorcnt(0);   // chunk kc landed in LDS
    __syncthreads();                          // publish buf to all waves
    if (kc + 1 < NCHUNK && threadIdx.x < 32)
      tdm_load_2d((unsigned)(size_t)&sB[buf ^ 1][0], bsrc + (size_t)(kc + 1) * 32,
                  KP, VPP, (unsigned long long)KP, 32, BN);

    const int k0 = kc * 32;
    if (kc + 2 < NCHUNK)                      // keep A stream warm in L0/L2
      __builtin_prefetch(arow[0] + k0 + 64, 0, 1);
    v16bf a[G2_MWT];
#pragma unroll
    for (int mi = 0; mi < G2_MWT; ++mi)
      a[mi] = load_a_frag(arow[mi] + k0, half);
#pragma unroll
    for (int ni = 0; ni < G2_NWT; ++ni) {
      int brow = wn * 64 + ni * 16 + n16;     // 0..255 within block N-tile
      FragU u;
      const uint4* p = reinterpret_cast<const uint4*>(&sB[buf][brow * 32 + half * 16]);
      u.q[0] = p[0];
      u.q[1] = p[1];
#pragma unroll
      for (int mi = 0; mi < G2_MWT; ++mi)
        acc[mi][ni] = __builtin_amdgcn_wmma_f32_16x16x32_bf16(
            false, a[mi], false, u.v, (short)0, acc[mi][ni], false, false);
    }
    __syncthreads();                          // all done reading buf
  }
#else
  // Fallback: cooperative synchronous global->LDS staging (single buffer).
  for (int kc = 0; kc < NCHUNK; ++kc) {
    const int k0 = kc * 32;
    __syncthreads();
    {
      int r = threadIdx.x;                    // one row per thread (256 rows)
      const uint4* src = reinterpret_cast<const uint4*>(bsrc + (size_t)r * KP + k0);
      uint4* dst = reinterpret_cast<uint4*>(&sB[0][r * 32]);
#pragma unroll
      for (int q = 0; q < 4; ++q) dst[q] = src[q];
    }
    __syncthreads();
    v16bf a[G2_MWT];
#pragma unroll
    for (int mi = 0; mi < G2_MWT; ++mi)
      a[mi] = load_a_frag(arow[mi] + k0, half);
#pragma unroll
    for (int ni = 0; ni < G2_NWT; ++ni) {
      int brow = wn * 64 + ni * 16 + n16;
      FragU u;
      const uint4* p = reinterpret_cast<const uint4*>(&sB[0][brow * 32 + half * 16]);
      u.q[0] = p[0];
      u.q[1] = p[1];
#pragma unroll
      for (int mi = 0; mi < G2_MWT; ++mi)
        acc[mi][ni] = __builtin_amdgcn_wmma_f32_16x16x32_bf16(
            false, a[mi], false, u.v, (short)0, acc[mi][ni], false, false);
    }
  }
#endif

  // Epilogue: bias + fp32 store. C/D layout: VGPR r -> row r + 8*half.
#pragma unroll
  for (int ni = 0; ni < G2_NWT; ++ni) {
    int col = n0 + wn * 64 + ni * 16 + n16;
    if (col < VOCAB) {
      float bias = b2[col];
#pragma unroll
      for (int mi = 0; mi < G2_MWT; ++mi) {
#pragma unroll
        for (int r = 0; r < 8; ++r) {
          int row = m0 + wm * 32 + mi * 16 + 8 * half + r;
          out[(size_t)row * VOCAB + col] = acc[mi][ni][r] + bias;
        }
      }
    }
  }
}

// ---------------------------------------------------------------------------
// Host-side launch. Workspace layout (bytes):
//   [0,        4 MiB)    Xb   bf16 2048x1024
//   [4 MiB,    8 MiB)    Hb   bf16 2048x1024
//   [8 MiB,   10 MiB)    W1t  bf16 1024x1024
//   [10 MiB, ~108.5 MiB) W2t  bf16 50432x1024
// ---------------------------------------------------------------------------
extern "C" void kernel_launch(void* const* d_in, const int* in_sizes, int n_in,
                              void* d_out, int out_size, void* d_ws, size_t ws_size,
                              hipStream_t stream) {
  const int*   tokens = (const int*)d_in[0];
  const float* embed  = (const float*)d_in[1];
  const float* W1     = (const float*)d_in[2];
  const float* b1     = (const float*)d_in[3];
  const float* W2     = (const float*)d_in[4];
  const float* b2     = (const float*)d_in[5];
  float* out = (float*)d_out;

  char* ws = (char*)d_ws;
  unsigned short* Xb  = (unsigned short*)(ws);
  unsigned short* Hb  = (unsigned short*)(ws + (size_t)4 * 1024 * 1024);
  unsigned short* W1t = (unsigned short*)(ws + (size_t)8 * 1024 * 1024);
  unsigned short* W2t = (unsigned short*)(ws + (size_t)10 * 1024 * 1024);

  // 1. gather + pack activations (bf16, K padded to 1024)
  pack_x_kernel<<<(M_ROWS * KP) / 256, 256, 0, stream>>>(tokens, embed, Xb);

  // 2. weight transposes (fp32 -> bf16, K-contiguous, zero padded)
  {
    dim3 blk(32, 8);
    dim3 g1(KP / 32, 1024 / 32);
    transpose_bf16_kernel<<<g1, blk, 0, stream>>>(W1, W1t, HID, HID, 1024);
    dim3 g2(KP / 32, VPP / 32);
    transpose_bf16_kernel<<<g2, blk, 0, stream>>>(W2, W2t, HID, VOCAB, VPP);
  }
  // 3. GEMM1 + bias + silu -> Hb
  {
    int ngroups_n = KP / (16 * NWT);                 // 16
    int waves = (M_ROWS / (16 * MWT)) * ngroups_n;   // 1024
    gemm1_silu_kernel<<<waves / 8, 256, 0, stream>>>(Xb, W1t, b1, Hb, ngroups_n);
  }
  // 4. GEMM2 + bias -> out (M-fastest grid: consecutive blocks share W2t in L2)
  {
    dim3 grid(M_ROWS / BM, VPP / BN);                // (32, 197)
    gemm2_kernel<<<grid, 256, 0, stream>>>(Hb, W2t, b2, out);
  }
}
